// EventBertSelfAttention_66915590472465
// MI455X (gfx1250) — matrix operand order repaired and checked by reading, not compiled, not run on hardware
//
#include <hip/hip_runtime.h>

typedef _Float16 v16h  __attribute__((ext_vector_type(16)));
typedef _Float16 v8h   __attribute__((ext_vector_type(8)));
typedef float    v8f   __attribute__((ext_vector_type(8)));
typedef float    f32x8 __attribute__((ext_vector_type(8)));

#define DEVINL __device__ __forceinline__

constexpr int Bc = 2, Sc = 2048, Hc = 1024, NHc = 16, DHc = 64;
constexpr float SOFTMAX_SCALE = 0.125f; // 1/sqrt(64), folded into Q

DEVINL v8f wmma_f16(v16h a, v16h b, v8f c) {
  // (neg_a, A, neg_b, B, c_mod, C, reuse_a, reuse_b)
  return __builtin_amdgcn_wmma_f32_16x16x32_f16(false, a, false, b, (short)0, c, false, false);
}

DEVINL v8f zero8() { v8f z = {0.f,0.f,0.f,0.f,0.f,0.f,0.f,0.f}; return z; }

// A-matrix 16x32 f16: lane holds row m = lane%16; K-granules of 8 selected by lane/16.
DEVINL v16h load_a_f16(const _Float16* row, int laneHi) {
  v8h lo = *(const v8h*)(row + laneHi * 8);
  v8h hi = *(const v8h*)(row + 16 + laneHi * 8);
  v16h r;
  #pragma unroll
  for (int i = 0; i < 8; ++i) { r[i] = lo[i]; r[i + 8] = hi[i]; }
  return r;
}

// B-matrix 32x16 f16: lane holds col n = lane%16; K-granules of 8 selected by lane/16.
DEVINL v16h load_b_f16(const _Float16* row, int laneHi) {
  v8h lo = *(const v8h*)(row + laneHi * 8);
  v8h hi = *(const v8h*)(row + 16 + laneHi * 8);
  v16h r;
  #pragma unroll
  for (int i = 0; i < 8; ++i) { r[i] = lo[i]; r[i + 8] = hi[i]; }
  return r;
}

// Same packings, f32 source converted on load.
DEVINL v16h load_a_f32cvt(const float* row, int laneHi) {
  f32x8 lo = *(const f32x8*)(row + laneHi * 8);
  f32x8 hi = *(const f32x8*)(row + 16 + laneHi * 8);
  v16h r;
  #pragma unroll
  for (int i = 0; i < 8; ++i) { r[i] = (_Float16)lo[i]; r[i + 8] = (_Float16)hi[i]; }
  return r;
}
DEVINL v16h load_b_f32cvt(const float* row, int laneHi) {
  f32x8 lo = *(const f32x8*)(row + laneHi * 8);
  f32x8 hi = *(const f32x8*)(row + 16 + laneHi * 8);
  v16h r;
  #pragma unroll
  for (int i = 0; i < 8; ++i) { r[i] = (_Float16)lo[i]; r[i + 8] = (_Float16)hi[i]; }
  return r;
}

// Issue one 16x32 f32 bias tile as 4 per-lane async b128 copies into LDS.
// Each lane: row = lane/2, 4-float chunk at col = (lane&1)*4 + c*8.
DEVINL void bias_async_issue(const float* bp, int k0, unsigned ldsBase, int lane) {
  const int row = lane >> 1;
  const int col = (lane & 1) * 4;
  const float* g = bp + (size_t)row * Sc + k0 + col;
  const unsigned l = ldsBase + (unsigned)(row * 32 + col) * 4u;
  #pragma unroll
  for (int c = 0; c < 4; ++c) {
    asm volatile("global_load_async_to_lds_b128 %0, %1, off"
                 :: "v"(l + (unsigned)c * 32u), "v"(g + c * 8)
                 : "memory");
  }
}

// ---------------------------------------------------------------------------
// Kernel 1: fused QKV projection.  One wave = one 16x16 output tile of Q,K,V.
// Q scaled by 1/sqrt(DH); Q,K stored [B,NH,S,DH] f16; V stored [B,NH,DH,S] f16
// (transposed via per-wave LDS tile) so P·V can consume it as row-major B.
// ---------------------------------------------------------------------------
__global__ __launch_bounds__(128) void qkv_proj_kernel(
    const float* __restrict__ X,
    const float* __restrict__ Wq, const float* __restrict__ bq,
    const float* __restrict__ Wk, const float* __restrict__ bk,
    const float* __restrict__ Wv, const float* __restrict__ bv,
    _Float16* __restrict__ qo, _Float16* __restrict__ ko, _Float16* __restrict__ vo)
{
  __shared__ alignas(16) _Float16 vtile[4][16][16]; // [wave][n][m] (transposed)
  const int wid = threadIdx.x >> 5;
  const int lane = threadIdx.x & 31;
  const int laneHi = lane >> 4, laneLo = lane & 15;
  const int t  = blockIdx.x * 4 + wid;     // 16384 wave-tiles total
  const int mt = t >> 6, nt = t & 63;      // 256 x 64 tiles
  const int m0 = mt * 16, n0 = nt * 16;

  const float* xrow = X  + (size_t)(m0 + laneLo) * Hc;
  const float* wqr  = Wq + (size_t)(n0 + laneLo) * Hc;
  const float* wkr  = Wk + (size_t)(n0 + laneLo) * Hc;
  const float* wvr  = Wv + (size_t)(n0 + laneLo) * Hc;

  v8f cq = zero8(), ck = zero8(), cv = zero8();
  for (int k = 0; k < Hc; k += 32) {
    v16h a   = load_a_f32cvt(xrow + k, laneHi);
    v16h bqm = load_b_f32cvt(wqr + k, laneHi);
    v16h bkm = load_b_f32cvt(wkr + k, laneHi);
    v16h bvm = load_b_f32cvt(wvr + k, laneHi);
    cq = wmma_f16(a, bqm, cq);
    ck = wmma_f16(a, bkm, ck);
    cv = wmma_f16(a, bvm, cv);
  }
  const float addq = bq[n0 + laneLo];
  const float addk = bk[n0 + laneLo];
  const float addv = bv[n0 + laneLo];

  const int b    = m0 >> 11;   // row / 2048
  const int s0   = m0 & 2047;
  const int head = n0 >> 6;
  const int d0   = n0 & 63;
  const size_t qkbase = (size_t)(b * NHc + head) * Sc * DHc;

  #pragma unroll
  for (int r = 0; r < 8; ++r) {
    const int s = s0 + r + laneHi * 8;                 // C/D row
    const size_t off = qkbase + (size_t)s * DHc + d0 + laneLo;
    qo[off] = (_Float16)((cq[r] + addq) * SOFTMAX_SCALE);
    ko[off] = (_Float16)(ck[r] + addk);
    vtile[wid][laneLo][r + laneHi * 8] = (_Float16)(cv[r] + addv);
  }
  __syncthreads();
  { // coalesced write of the transposed V tile: 2 lanes per d-row, b128 each
    const int dr = lane >> 1, half = lane & 1;
    v8h rowv = *(const v8h*)&vtile[wid][dr][half * 8];
    const size_t off = ((size_t)(b * NHc + head) * DHc + d0 + dr) * Sc + s0 + half * 8;
    *(v8h*)(vo + off) = rowv;
  }
}

// ---------------------------------------------------------------------------
// Kernel 2: flash attention.  One wave owns a 16-row Q tile of one (b, head)
// and streams 64 key-tiles of 32 with online softmax.  Batch is innermost in
// the wave-tile order so both batches hit the same ALiBi bias rows back-to-
// back (L2/WGP$ reuse of the 256 MB stream).  Bias tiles are double-buffered
// through LDS with GLOBAL_LOAD_ASYNC_TO_LDS_B128 (ASYNCcnt pipelining).
// ---------------------------------------------------------------------------
__global__ __launch_bounds__(128) void flash_attn_kernel(
    const _Float16* __restrict__ qm, const _Float16* __restrict__ km,
    const _Float16* __restrict__ vtm, const float* __restrict__ bias,
    float* __restrict__ out)
{
  __shared__ alignas(16) float   btile[4][2][16][32]; // bias double buffer, 16 KB
  __shared__ alignas(16) _Float16 ptile[4][16][32];   // P staging, 4 KB
  const int wid  = threadIdx.x >> 5;
  const int lane = threadIdx.x & 31;
  const int laneHi = lane >> 4, laneLo = lane & 15;
  const int t    = blockIdx.x * 4 + wid;   // 4096 wave-tiles total
  const int head = t >> 8;                 // batch innermost for bias reuse
  const int q0   = ((t >> 1) & 127) * 16;
  const int b    = t & 1;
  const int bh   = b * NHc + head;

  const _Float16* qp = qm  + (size_t)bh * Sc * DHc;
  const _Float16* kp = km  + (size_t)bh * Sc * DHc;
  const _Float16* vp = vtm + (size_t)bh * DHc * Sc;   // [DH][S]
  const float*    bp = bias + (size_t)head * Sc * Sc + (size_t)q0 * Sc;

  const unsigned ldsB0 = (unsigned)(uintptr_t)&btile[wid][0][0][0];
  const unsigned ldsB1 = (unsigned)(uintptr_t)&btile[wid][1][0][0];

  // Q tile is A-resident for the whole loop (DH=64 -> two K=32 A tiles).
  const _Float16* qrow = qp + (size_t)(q0 + laneLo) * DHc;
  const v16h aq0 = load_a_f16(qrow, laneHi);
  const v16h aq1 = load_a_f16(qrow + 32, laneHi);

  float mrow[8], lrow[8];
  v8f acc[4];
  #pragma unroll
  for (int r = 0; r < 8; ++r) { mrow[r] = -1e30f; lrow[r] = 0.f; }
  #pragma unroll
  for (int dt = 0; dt < 4; ++dt) acc[dt] = zero8();

  bias_async_issue(bp, 0, ldsB0, lane);    // prime buffer 0

  for (int kt = 0; kt < Sc / 32; ++kt) {
    const int k0 = kt * 32;
    if (kt + 1 < Sc / 32) {                // prefetch next bias tile, then
      bias_async_issue(bp, k0 + 32, (kt & 1) ? ldsB0 : ldsB1, lane);
      asm volatile("s_wait_asynccnt 0x4" ::: "memory");  // buffer kt landed
    } else {
      asm volatile("s_wait_asynccnt 0x0" ::: "memory");
    }

    // S = Q · K^T : B operand is rows of K packed over d (no transpose needed)
    const _Float16* kr0 = kp + (size_t)(k0 + laneLo) * DHc;
    const _Float16* kr1 = kp + (size_t)(k0 + 16 + laneLo) * DHc;
    v8f sc0 = wmma_f16(aq0, load_b_f16(kr0,      laneHi), zero8());
    sc0     = wmma_f16(aq1, load_b_f16(kr0 + 32, laneHi), sc0);
    v8f sc1 = wmma_f16(aq0, load_b_f16(kr1,      laneHi), zero8());
    sc1     = wmma_f16(aq1, load_b_f16(kr1 + 32, laneHi), sc1);

    const float* bb = &btile[wid][kt & 1][0][0];
    float p0[8], p1[8];
    #pragma unroll
    for (int r = 0; r < 8; ++r) {          // ALiBi bias from LDS
      const int brow = (r + laneHi * 8) * 32 + laneLo;
      sc0[r] += bb[brow];
      sc1[r] += bb[brow + 16];
    }
    #pragma unroll
    for (int r = 0; r < 8; ++r) {          // online softmax, row = 16-lane half
      float v = fmaxf(sc0[r], sc1[r]);
      v = fmaxf(v, __shfl_xor(v, 1, 32));
      v = fmaxf(v, __shfl_xor(v, 2, 32));
      v = fmaxf(v, __shfl_xor(v, 4, 32));
      v = fmaxf(v, __shfl_xor(v, 8, 32));
      const float mn = fmaxf(mrow[r], v);
      const float alpha = __expf(mrow[r] - mn);
      mrow[r] = mn;
      p0[r] = __expf(sc0[r] - mn);
      p1[r] = __expf(sc1[r] - mn);
      float rs = p0[r] + p1[r];
      rs += __shfl_xor(rs, 1, 32);
      rs += __shfl_xor(rs, 2, 32);
      rs += __shfl_xor(rs, 4, 32);
      rs += __shfl_xor(rs, 8, 32);
      lrow[r] = lrow[r] * alpha + rs;
      #pragma unroll
      for (int dt = 0; dt < 4; ++dt) acc[dt][r] *= alpha;
    }
    // C/D layout -> A layout transpose of P through LDS
    #pragma unroll
    for (int r = 0; r < 8; ++r) {
      ptile[wid][r + laneHi * 8][laneLo]      = (_Float16)p0[r];
      ptile[wid][r + laneHi * 8][16 + laneLo] = (_Float16)p1[r];
    }
    __syncthreads();
    const v16h ap = load_a_f16(&ptile[wid][laneLo][0], laneHi);
    __syncthreads();
    // ctx += P · V : B operand is rows of V^T packed over k
    #pragma unroll
    for (int dt = 0; dt < 4; ++dt) {
      const _Float16* vrow = vp + (size_t)(dt * 16 + laneLo) * Sc + k0;
      acc[dt] = wmma_f16(ap, load_b_f16(vrow, laneHi), acc[dt]);
    }
  }

  #pragma unroll
  for (int r = 0; r < 8; ++r) {
    const float inv = 1.0f / lrow[r];
    const int s = q0 + r + laneHi * 8;
    #pragma unroll
    for (int dt = 0; dt < 4; ++dt) {
      out[((size_t)b * Sc + s) * Hc + head * DHc + dt * 16 + laneLo] = acc[dt][r] * inv;
    }
  }
}

extern "C" void kernel_launch(void* const* d_in, const int* in_sizes, int n_in,
                              void* d_out, int out_size, void* d_ws, size_t ws_size,
                              hipStream_t stream) {
  (void)in_sizes; (void)n_in; (void)out_size; (void)ws_size;
  const float* X    = (const float*)d_in[0];
  const float* bias = (const float*)d_in[1];
  const float* Wq   = (const float*)d_in[2];
  const float* bq   = (const float*)d_in[3];
  const float* Wk   = (const float*)d_in[4];
  const float* bk   = (const float*)d_in[5];
  const float* Wv   = (const float*)d_in[6];
  const float* bv   = (const float*)d_in[7];
  float* out = (float*)d_out;

  const size_t per = (size_t)Bc * NHc * Sc * DHc;  // 4,194,304 f16 each
  _Float16* qws = (_Float16*)d_ws;
  _Float16* kws = qws + per;
  _Float16* vws = kws + per;

  const int qkv_tiles  = (Bc * Sc / 16) * (Hc / 16);   // 16384 wave-tiles
  qkv_proj_kernel<<<qkv_tiles / 4, 128, 0, stream>>>(X, Wq, bq, Wk, bk, Wv, bv,
                                                     qws, kws, vws);

  const int attn_tiles = Bc * NHc * (Sc / 16);         // 4096 wave-tiles
  flash_attn_kernel<<<attn_tiles / 4, 128, 0, stream>>>(qws, kws, vws, bias, out);
}
